// Tranformer_10179072491955
// MI455X (gfx1250) — matrix operand order repaired and checked by reading, not compiled, not run on hardware
//
#include <hip/hip_runtime.h>

// ---------------------------------------------------------------------------
// MI455X (gfx1250) transformer forward.  All matmuls run through
// v_wmma_f32_16x16x32_bf16 (wave32, f32 accumulate).  ~206 GFLOP total,
// compute-bound (bf16 weights ~36MB stay resident in the 192MB L2), so the
// pipeline is organized around feeding the WMMA units from double-buffered
// LDS tiles, using the CDNA5 async global->LDS path when available.
// ---------------------------------------------------------------------------

typedef __attribute__((ext_vector_type(16))) __bf16 v16bf;
typedef __attribute__((ext_vector_type(8)))  float  v8f;

union Frag { uint4 q[2]; v16bf v; };

__device__ __forceinline__ unsigned short f2bf(float f) {
  unsigned int u = __float_as_uint(f);
  u += 0x7FFFu + ((u >> 16) & 1u);        // round-to-nearest-even
  return (unsigned short)(u >> 16);
}

// -------- CDNA5 async global->LDS copy (ASYNCcnt-tracked), if exposed ------
#if defined(__AMDGCN__) && __has_builtin(__builtin_amdgcn_global_load_async_to_lds_b128)
#define HAS_ASYNC_LDS 1
#else
#define HAS_ASYNC_LDS 0
#endif

#if HAS_ASYNC_LDS
typedef __attribute__((__vector_size__(4 * sizeof(int)))) int v4i;
typedef __attribute__((address_space(3))) v4i lds_v4i;   // LDS-qualified int4
__device__ __forceinline__ void async_copy16(const unsigned short* g,
                                             unsigned short* l) {
  // param0: generic int4* (per compiler diagnostic); param1: LDS int4*
  // (AS3 converts implicitly to generic on AMDGPU if param1 is generic).
  __builtin_amdgcn_global_load_async_to_lds_b128((v4i*)g, (lds_v4i*)l, 0, 0);
}
__device__ __forceinline__ void async_wait0() {
#if __has_builtin(__builtin_amdgcn_s_wait_asynccnt)
  __builtin_amdgcn_s_wait_asynccnt(0);
#else
  asm volatile("s_wait_asynccnt 0x0" ::: "memory");
#endif
}
#endif

// ---------------------------------------------------------------------------
// f32 -> bf16 conversion
// ---------------------------------------------------------------------------
__global__ __launch_bounds__(256) void cvt_bf16(const float* __restrict__ s,
                                                unsigned short* __restrict__ d,
                                                int n) {
  int i = blockIdx.x * 256 + threadIdx.x;
  if (i < n) d[i] = f2bf(s[i]);
}

// ---------------------------------------------------------------------------
// C[M,N] = act( (A[MxK] * W[NxK]^T + bias) * alpha )
// A, W bf16 row-major (K-contiguous).  128x128 block tile, 8 waves,
// each wave 64x32 (4 M-tiles x 2 N-tiles of 16x16), K-step 32,
// double-buffered LDS with software-pipelined tile fetch.
// ---------------------------------------------------------------------------
#define GS 40   // LDS row stride (ushorts): 80B, 16B aligned, conflict-free

__global__ __launch_bounds__(256) void gemm_bf16(
    const unsigned short* __restrict__ A,
    const unsigned short* __restrict__ W,
    const float* __restrict__ bias,
    float* __restrict__ Cf,
    unsigned short* __restrict__ Cb,
    int M, int N, int K, float alpha, int relu)
{
  __shared__ unsigned short lA[2][128 * GS];
  __shared__ unsigned short lB[2][128 * GS];
  const int tid  = threadIdx.x;
  const int wid  = tid >> 5;
  const int lane = tid & 31;
  const int hl   = lane >> 4;      // which 16-lane half
  const int l16  = lane & 15;
  const int wm   = wid & 1;        // 2 wave rows
  const int wn   = wid >> 1;       // 4 wave cols
  const long bm  = (long)blockIdx.y * 128;
  const long bn  = (long)blockIdx.x * 128;

  // this thread's two staging slots (128x32 tile = 512 x 8-bf16 vectors)
  const int r0 = tid >> 2,          c80 = (tid & 3) * 8;
  const int r1 = (tid + 256) >> 2,  c81 = ((tid + 256) & 3) * 8;
  const unsigned short* ga0 = A + (bm + r0) * (long)K + c80;
  const unsigned short* ga1 = A + (bm + r1) * (long)K + c81;
  const unsigned short* gw0 = W + (bn + r0) * (long)K + c80;
  const unsigned short* gw1 = W + (bn + r1) * (long)K + c81;
  const int sl0 = r0 * GS + c80;
  const int sl1 = r1 * GS + c81;

#if HAS_ASYNC_LDS
  auto issue = [&](int buf, int k0) {
    async_copy16(ga0 + k0, lA[buf] + sl0);
    async_copy16(ga1 + k0, lA[buf] + sl1);
    async_copy16(gw0 + k0, lB[buf] + sl0);
    async_copy16(gw1 + k0, lB[buf] + sl1);
  };
  auto commit = [&](int) { async_wait0(); };
#else
  uint4 ra0, ra1, rb0, rb1;
  auto issue = [&](int, int k0) {
    ra0 = *(const uint4*)(ga0 + k0);
    ra1 = *(const uint4*)(ga1 + k0);
    rb0 = *(const uint4*)(gw0 + k0);
    rb1 = *(const uint4*)(gw1 + k0);
  };
  auto commit = [&](int buf) {
    *(uint4*)(lA[buf] + sl0) = ra0;
    *(uint4*)(lA[buf] + sl1) = ra1;
    *(uint4*)(lB[buf] + sl0) = rb0;
    *(uint4*)(lB[buf] + sl1) = rb1;
  };
#endif

  v8f acc[4][2];
  for (int mt = 0; mt < 4; ++mt)
    for (int nt = 0; nt < 2; ++nt)
      for (int r = 0; r < 8; ++r) acc[mt][nt][r] = 0.f;

  issue(0, 0);
  commit(0);
  __syncthreads();

  int cur = 0;
  for (int k0 = 0; k0 < K; k0 += 32) {
    const bool last = (k0 + 32 >= K);
    if (!last) issue(cur ^ 1, k0 + 32);   // prefetch next tile (overlaps WMMA)

    Frag a[4], b[2];
#pragma unroll
    for (int mt = 0; mt < 4; ++mt) {      // A frag: K = hl*16+[0..15]
      const unsigned short* p = lA[cur] + (wm * 64 + mt * 16 + l16) * GS + hl * 16;
      a[mt].q[0] = *(const uint4*)(p);
      a[mt].q[1] = *(const uint4*)(p + 8);
    }
#pragma unroll
    for (int nt = 0; nt < 2; ++nt) {      // B frag: K = hl*16+[0..15]
      const unsigned short* p = lB[cur] + (wn * 32 + nt * 16 + l16) * GS + hl * 16;
      b[nt].q[0] = *(const uint4*)(p);
      b[nt].q[1] = *(const uint4*)(p + 8);
    }
#pragma unroll
    for (int mt = 0; mt < 4; ++mt)
#pragma unroll
      for (int nt = 0; nt < 2; ++nt)
        acc[mt][nt] = __builtin_amdgcn_wmma_f32_16x16x32_bf16(
            false, a[mt].v, false, b[nt].v, (short)0, acc[mt][nt], false, false);

    if (!last) commit(cur ^ 1);
    __syncthreads();
    cur ^= 1;
  }

#pragma unroll
  for (int mt = 0; mt < 4; ++mt) {
#pragma unroll
    for (int nt = 0; nt < 2; ++nt) {
      long col = bn + wn * 32 + nt * 16 + l16;
      float bvv = bias ? bias[col] : 0.f;
#pragma unroll
      for (int r = 0; r < 8; ++r) {       // C layout: row = hl*8 + r
        long row = bm + wm * 64 + mt * 16 + hl * 8 + r;
        float val = (acc[mt][nt][r] + bvv) * alpha;
        if (relu) val = fmaxf(val, 0.f);
        long idx = row * (long)N + col;
        if (Cf) Cf[idx] = val;
        if (Cb) Cb[idx] = f2bf(val);
      }
    }
  }
}

// ---------------------------------------------------------------------------
// Flash attention: one (b,h) and 64 q-rows per block, 4 waves, online softmax.
// Q/K/V bf16 (B,L,E) with E = H*64; q pre-scaled by 1/sqrt(HD).
// ---------------------------------------------------------------------------
#define AS 72   // LDS row stride (ushorts): 144B, 16B aligned

__global__ __launch_bounds__(128) void attn_flash(
    const unsigned short* __restrict__ Qm,
    const unsigned short* __restrict__ Km,
    const unsigned short* __restrict__ Vm,
    const float* __restrict__ kmask,
    unsigned short* __restrict__ Om)
{
  const int L = 1024, E = 512;
  __shared__ unsigned short q_l[64 * AS];
  __shared__ unsigned short k_l[64 * AS];
  __shared__ unsigned short v_l[64 * AS];       // transposed: [hd][key]
  __shared__ unsigned short p_l[4][16 * AS];    // wave-private P staging

  const int tid  = threadIdx.x;
  const int w    = tid >> 5;
  const int lane = tid & 31;
  const int hl   = lane >> 4;
  const int l16  = lane & 15;
  const int qb   = blockIdx.x * 64;
  const int h    = blockIdx.y;
  const int b    = blockIdx.z;
  const long base = ((long)b * L) * E + h * 64;

  // stage Q tile (64 x 64)
#if HAS_ASYNC_LDS
#pragma unroll
  for (int i = 0; i < 4; ++i) {
    int vv = tid + i * 128;
    int r = vv >> 3, c8 = (vv & 7) * 8;
    async_copy16(Qm + base + (long)(qb + r) * E + c8, q_l + r * AS + c8);
  }
  async_wait0();
#else
  {
    uint4 qr[4];
#pragma unroll
    for (int i = 0; i < 4; ++i) {
      int vv = tid + i * 128;
      int r = vv >> 3, c8 = (vv & 7) * 8;
      qr[i] = *(const uint4*)(Qm + base + (long)(qb + r) * E + c8);
    }
#pragma unroll
    for (int i = 0; i < 4; ++i) {
      int vv = tid + i * 128;
      int r = vv >> 3, c8 = (vv & 7) * 8;
      *(uint4*)(q_l + r * AS + c8) = qr[i];
    }
  }
#endif
  __syncthreads();

  Frag qa[2];
#pragma unroll
  for (int st = 0; st < 2; ++st) {
    const unsigned short* p = q_l + (w * 16 + l16) * AS + st * 32 + hl * 16;
    qa[st].q[0] = *(const uint4*)(p);
    qa[st].q[1] = *(const uint4*)(p + 8);
  }

  v8f acco[4];
  float rm[8], rl[8];
#pragma unroll
  for (int nt = 0; nt < 4; ++nt)
    for (int r = 0; r < 8; ++r) acco[nt][r] = 0.f;
#pragma unroll
  for (int r = 0; r < 8; ++r) { rm[r] = -1e30f; rl[r] = 0.f; }

  for (int c = 0; c < L; c += 64) {
    __syncthreads();
    // ---- stage K chunk [key][hd] and V chunk transposed [hd][key] ----
#if HAS_ASYNC_LDS
#pragma unroll
    for (int i = 0; i < 4; ++i) {
      int vv = tid + i * 128;
      int r = vv >> 3, c8 = (vv & 7) * 8;
      async_copy16(Km + base + (long)(c + r) * E + c8, k_l + r * AS + c8);
    }
#else
    uint4 kr[4];
#pragma unroll
    for (int i = 0; i < 4; ++i) {
      int vv = tid + i * 128;
      int r = vv >> 3, c8 = (vv & 7) * 8;
      kr[i] = *(const uint4*)(Km + base + (long)(c + r) * E + c8);
    }
#endif
    unsigned int vr[16];
#pragma unroll
    for (int i = 0; i < 16; ++i) {        // 2048 dword loads cover 64x64
      int e2 = i * 128 + tid;
      int key = e2 >> 5, hd2 = (e2 & 31) * 2;
      vr[i] = *(const unsigned int*)(Vm + base + (long)(c + key) * E + hd2);
    }
#if !HAS_ASYNC_LDS
#pragma unroll
    for (int i = 0; i < 4; ++i) {
      int vv = tid + i * 128;
      int r = vv >> 3, c8 = (vv & 7) * 8;
      *(uint4*)(k_l + r * AS + c8) = kr[i];
    }
#endif
#pragma unroll
    for (int i = 0; i < 16; ++i) {
      int e2 = i * 128 + tid;
      int key = e2 >> 5, hd2 = (e2 & 31) * 2;
      v_l[hd2 * AS + key]       = (unsigned short)(vr[i] & 0xFFFFu);
      v_l[(hd2 + 1) * AS + key] = (unsigned short)(vr[i] >> 16);
    }
#if HAS_ASYNC_LDS
    async_wait0();
#endif
    __syncthreads();

    // ---- S = Q * K^T : 16 x 64 per wave ----
    v8f s[4];
#pragma unroll
    for (int nt = 0; nt < 4; ++nt) {
      for (int r = 0; r < 8; ++r) s[nt][r] = 0.f;
#pragma unroll
      for (int st = 0; st < 2; ++st) {
        Frag kb;
        const unsigned short* p = k_l + (nt * 16 + l16) * AS + st * 32 + hl * 16;
        kb.q[0] = *(const uint4*)(p);
        kb.q[1] = *(const uint4*)(p + 8);
        s[nt] = __builtin_amdgcn_wmma_f32_16x16x32_bf16(
            false, qa[st].v, false, kb.v, (short)0, s[nt], false, false);
      }
      float km = kmask[(long)b * L + c + nt * 16 + l16];
      if (km == 0.f)
        for (int r = 0; r < 8; ++r) s[nt][r] = -1e30f;
    }

    // ---- online softmax (row m = hl*8 + r lives in one 16-lane half) ----
#pragma unroll
    for (int r = 0; r < 8; ++r) {
      float mx = fmaxf(fmaxf(s[0][r], s[1][r]), fmaxf(s[2][r], s[3][r]));
      mx = fmaxf(mx, __shfl_xor(mx, 1, 32));
      mx = fmaxf(mx, __shfl_xor(mx, 2, 32));
      mx = fmaxf(mx, __shfl_xor(mx, 4, 32));
      mx = fmaxf(mx, __shfl_xor(mx, 8, 32));
      float mn  = fmaxf(rm[r], mx);
      float fac = __expf(rm[r] - mn);
      rm[r] = mn;
      float ps = 0.f;
#pragma unroll
      for (int nt = 0; nt < 4; ++nt) {
        float pv = __expf(s[nt][r] - mn);
        s[nt][r] = pv;
        ps += pv;
      }
      ps += __shfl_xor(ps, 1, 32);
      ps += __shfl_xor(ps, 2, 32);
      ps += __shfl_xor(ps, 4, 32);
      ps += __shfl_xor(ps, 8, 32);
      rl[r] = rl[r] * fac + ps;
#pragma unroll
      for (int nt = 0; nt < 4; ++nt) acco[nt][r] *= fac;
    }

    // ---- stage P as bf16 (wave-private; LDS ops in-order per wave) ----
#pragma unroll
    for (int nt = 0; nt < 4; ++nt)
#pragma unroll
      for (int r = 0; r < 8; ++r)
        p_l[w][(hl * 8 + r) * AS + nt * 16 + l16] = f2bf(s[nt][r]);

    // ---- O += P * V ----
#pragma unroll
    for (int st = 0; st < 2; ++st) {
      Frag pa;
      const unsigned short* p = p_l[w] + l16 * AS + st * 32 + hl * 16;
      pa.q[0] = *(const uint4*)(p);
      pa.q[1] = *(const uint4*)(p + 8);
#pragma unroll
      for (int nt = 0; nt < 4; ++nt) {
        Frag vb;
        const unsigned short* pv = v_l + (nt * 16 + l16) * AS + st * 32 + hl * 16;
        vb.q[0] = *(const uint4*)(pv);
        vb.q[1] = *(const uint4*)(pv + 8);
        acco[nt] = __builtin_amdgcn_wmma_f32_16x16x32_bf16(
            false, pa.v, false, vb.v, (short)0, acco[nt], false, false);
      }
    }
  }

#pragma unroll
  for (int nt = 0; nt < 4; ++nt) {
#pragma unroll
    for (int r = 0; r < 8; ++r) {
      long qrow = qb + w * 16 + hl * 8 + r;
      int hd = nt * 16 + l16;
      Om[base + qrow * (long)E + hd] = f2bf(acco[nt][r] / rl[r]);
    }
  }
}

// ---------------------------------------------------------------------------
// x = LayerNorm(gin * qm + x); also emit bf16 copy.  One row (D=512)/block.
// ---------------------------------------------------------------------------
__global__ __launch_bounds__(256) void ln_res(
    const float* __restrict__ gin, const float* __restrict__ qm,
    const float* __restrict__ gamma, const float* __restrict__ beta,
    float* __restrict__ x, unsigned short* __restrict__ xb)
{
  __shared__ float red[256];
  __shared__ float mean_s, rstd_s;
  const int row = blockIdx.x;
  const int t = threadIdx.x;
  const long o = (long)row * 512;
  float qv = qm[row];
  float v0 = gin[o + t] * qv + x[o + t];
  float v1 = gin[o + 256 + t] * qv + x[o + 256 + t];
  red[t] = v0 + v1;
  __syncthreads();
  for (int s = 128; s > 0; s >>= 1) {
    if (t < s) red[t] += red[t + s];
    __syncthreads();
  }
  if (t == 0) mean_s = red[0] * (1.f / 512.f);
  __syncthreads();
  float mean = mean_s;
  float d0 = v0 - mean, d1 = v1 - mean;
  red[t] = d0 * d0 + d1 * d1;
  __syncthreads();
  for (int s = 128; s > 0; s >>= 1) {
    if (t < s) red[t] += red[t + s];
    __syncthreads();
  }
  if (t == 0) rstd_s = rsqrtf(red[0] * (1.f / 512.f) + 1e-5f);
  __syncthreads();
  float rs = rstd_s;
  float y0 = d0 * rs * gamma[t] + beta[t];
  float y1 = d1 * rs * gamma[t + 256] + beta[t + 256];
  x[o + t] = y0;        xb[o + t] = f2bf(y0);
  x[o + 256 + t] = y1;  xb[o + 256 + t] = f2bf(y1);
}

// ---------------------------------------------------------------------------
// Host orchestration
// ---------------------------------------------------------------------------
extern "C" void kernel_launch(void* const* d_in, const int* in_sizes, int n_in,
                              void* d_out, int out_size, void* d_ws, size_t ws_size,
                              hipStream_t stream)
{
  (void)in_sizes; (void)n_in; (void)out_size; (void)ws_size;
  const int Bc = 4, Lc = 1024, Dc = 512, Hc = 8, Ec = 512, Fc = 2048, LAY = 6;
  const long M = (long)Bc * Lc;  // 4096

  const float* query = (const float*)d_in[0];
  const float* key   = (const float*)d_in[1];
  const float* qmask = (const float*)d_in[2];
  const float* kmask = (const float*)d_in[3];
  const float* Wq = (const float*)d_in[4];
  const float* bq = (const float*)d_in[5];
  const float* Wk = (const float*)d_in[6];
  const float* bk = (const float*)d_in[7];
  const float* Wv = (const float*)d_in[8];
  const float* bv = (const float*)d_in[9];
  const float* Wo = (const float*)d_in[10];
  const float* bo = (const float*)d_in[11];
  const float* W1 = (const float*)d_in[12];
  const float* b1 = (const float*)d_in[13];
  const float* W2 = (const float*)d_in[14];
  const float* b2 = (const float*)d_in[15];
  const float* g1 = (const float*)d_in[16];
  const float* be1 = (const float*)d_in[17];
  const float* g2 = (const float*)d_in[18];
  const float* be2 = (const float*)d_in[19];

  char* wp = (char*)d_ws;
  auto carve = [&](size_t bytes) -> void* {
    void* r = (void*)wp;
    wp += (bytes + 255) & ~(size_t)255;
    return r;
  };
  unsigned short* wqb  = (unsigned short*)carve((size_t)LAY * Ec * Dc * 2);
  unsigned short* wkb  = (unsigned short*)carve((size_t)LAY * Ec * Dc * 2);
  unsigned short* wvb  = (unsigned short*)carve((size_t)LAY * Ec * Dc * 2);
  unsigned short* wob  = (unsigned short*)carve((size_t)LAY * Dc * Ec * 2);
  unsigned short* w1b  = (unsigned short*)carve((size_t)LAY * Fc * Dc * 2);
  unsigned short* w2b  = (unsigned short*)carve((size_t)LAY * Dc * Fc * 2);
  unsigned short* keyb = (unsigned short*)carve((size_t)M * Dc * 2);
  float*          xf   = (float*)         carve((size_t)M * Dc * 4);
  unsigned short* xb   = (unsigned short*)carve((size_t)M * Dc * 2);
  unsigned short* qb2  = (unsigned short*)carve((size_t)M * Ec * 2);
  unsigned short* kb2  = (unsigned short*)carve((size_t)M * Ec * 2);
  unsigned short* vb2  = (unsigned short*)carve((size_t)M * Ec * 2);
  unsigned short* atb  = (unsigned short*)carve((size_t)M * Ec * 2);
  float*          tmpf = (float*)         carve((size_t)M * Dc * 4);
  unsigned short* hb   = (unsigned short*)carve((size_t)M * Fc * 2);

  auto cvt = [&](const float* s, unsigned short* d, long n) {
    cvt_bf16<<<dim3((unsigned)((n + 255) / 256)), dim3(256), 0, stream>>>(s, d, (int)n);
  };
  auto gemm = [&](const unsigned short* A, const unsigned short* Wt,
                  const float* bias, float* Cf, unsigned short* Cb,
                  int Mm, int Nn, int Kk, float alpha, int relu) {
    dim3 g((unsigned)(Nn / 128), (unsigned)(Mm / 128));
    gemm_bf16<<<g, dim3(256), 0, stream>>>(A, Wt, bias, Cf, Cb, Mm, Nn, Kk, alpha, relu);
  };

  // one-time conversions (weights stay resident in L2 as bf16: ~36 MB)
  cvt(Wq, wqb, (long)LAY * Ec * Dc);
  cvt(Wk, wkb, (long)LAY * Ec * Dc);
  cvt(Wv, wvb, (long)LAY * Ec * Dc);
  cvt(Wo, wob, (long)LAY * Dc * Ec);
  cvt(W1, w1b, (long)LAY * Fc * Dc);
  cvt(W2, w2b, (long)LAY * Dc * Fc);
  cvt(key, keyb, M * Dc);
  cvt(query, xb, M * Dc);
  (void)hipMemcpyAsync(xf, query, (size_t)M * Dc * sizeof(float),
                       hipMemcpyDeviceToDevice, stream);

  const float qscale = 0.125f;  // 1/sqrt(HD=64), folded into q projection

  for (int i = 0; i < LAY; ++i) {
    const unsigned short* wq = wqb + (long)i * Ec * Dc;
    const unsigned short* wk = wkb + (long)i * Ec * Dc;
    const unsigned short* wv = wvb + (long)i * Ec * Dc;
    const unsigned short* wo = wob + (long)i * Dc * Ec;
    const unsigned short* w1 = w1b + (long)i * Fc * Dc;
    const unsigned short* w2 = w2b + (long)i * Dc * Fc;

    gemm(xb,   wq, bq + (long)i * Ec, nullptr, qb2, (int)M, Ec, Dc, qscale, 0);
    gemm(keyb, wk, bk + (long)i * Ec, nullptr, kb2, (int)M, Ec, Dc, 1.f, 0);
    gemm(keyb, wv, bv + (long)i * Ec, nullptr, vb2, (int)M, Ec, Dc, 1.f, 0);

    attn_flash<<<dim3(Lc / 64, Hc, Bc), dim3(128), 0, stream>>>(
        qb2, kb2, vb2, kmask, atb);

    gemm(atb, wo, bo + (long)i * Dc, tmpf, nullptr, (int)M, Dc, Ec, 1.f, 0);
    ln_res<<<dim3((unsigned)M), dim3(256), 0, stream>>>(
        tmpf, qmask, g1 + (long)i * Dc, be1 + (long)i * Dc, xf, xb);

    gemm(xb, w1, b1 + (long)i * Fc, nullptr, hb, (int)M, Fc, Dc, 1.f, 1);
    gemm(hb, w2, b2 + (long)i * Dc, tmpf, nullptr, (int)M, Dc, Fc, 1.f, 0);
    ln_res<<<dim3((unsigned)M), dim3(256), 0, stream>>>(
        tmpf, qmask, g2 + (long)i * Dc, be2 + (long)i * Dc, xf, xb);
  }

  (void)hipMemcpyAsync(d_out, xf, (size_t)M * Dc * sizeof(float),
                       hipMemcpyDeviceToDevice, stream);
}
